// JambaMoE_67242007986397
// MI455X (gfx1250) — matrix operand order repaired and checked by reading, not compiled, not run on hardware
//
#include <hip/hip_runtime.h>
#include <hip/hip_bf16.h>

// MoE (Jamba-style) for MI455X / gfx1250, wave32 + v_wmma_f32_16x16x32_bf16.
// Top-2-sparse dispatch (deterministic, atomics-free), bf16 tensor-core GEMMs
// with on-the-fly fp32->bf16 conversion during LDS staging.
//
// Scratch layout (needs ~129 MB of d_ws):
//   top_ids[T][2] i32 | top_w[T][2] f32 | list_tok[2T] i32 | slot_w[2T] f32 |
//   pos[T][2] i32 | counts[8] | bases[8] | h[2T][I] bf16 | Y[2T][H] f32

typedef __attribute__((ext_vector_type(16))) __bf16 v16bf;
typedef __attribute__((ext_vector_type(8)))  float  v8f;

static constexpr int T = 4096, H = 2048, E = 8, I = 4096;
static constexpr int TWOI = 2 * I;

static constexpr size_t OFF_TOPID = 0;
static constexpr size_t OFF_TOPW  = OFF_TOPID + (size_t)T * 2 * 4;
static constexpr size_t OFF_LIST  = OFF_TOPW  + (size_t)T * 2 * 4;
static constexpr size_t OFF_SLOTW = OFF_LIST  + (size_t)2 * T * 4;
static constexpr size_t OFF_POS   = OFF_SLOTW + (size_t)2 * T * 4;
static constexpr size_t OFF_CNT   = OFF_POS   + (size_t)T * 2 * 4;
static constexpr size_t OFF_BASE  = OFF_CNT + 128;
static constexpr size_t OFF_H     = 262144;                       // 256 KB aligned
static constexpr size_t OFF_Y     = OFF_H + (size_t)2 * T * I * 2;

// ---------------------------------------------------------------- router ----
__launch_bounds__(256)
__global__ void router_kernel(const float* __restrict__ x,
                              const float* __restrict__ wr,
                              int* __restrict__ top_ids,
                              float* __restrict__ top_w) {
  const int wave = threadIdx.x >> 5;
  const int lane = threadIdx.x & 31;
  const int token = blockIdx.x * 8 + wave;
  const float* xr = x + (size_t)token * H;

  float acc[E];
#pragma unroll
  for (int e = 0; e < E; ++e) acc[e] = 0.f;
  for (int h = lane; h < H; h += 32) {
    const float xv = xr[h];
#pragma unroll
    for (int e = 0; e < E; ++e) acc[e] += xv * wr[e * H + h];
  }
#pragma unroll
  for (int e = 0; e < E; ++e) {
#pragma unroll
    for (int off = 16; off > 0; off >>= 1)
      acc[e] += __shfl_xor(acc[e], off, 32);
  }
  if (lane == 0) {
    float m = acc[0];
#pragma unroll
    for (int e = 1; e < E; ++e) m = fmaxf(m, acc[e]);
    float p[E], s = 0.f;
#pragma unroll
    for (int e = 0; e < E; ++e) { p[e] = __expf(acc[e] - m); s += p[e]; }
    const float inv = 1.f / s;
    int i0 = 0; float p0 = p[0];
#pragma unroll
    for (int e = 1; e < E; ++e) if (p[e] > p0) { p0 = p[e]; i0 = e; }
    int i1 = -1; float p1 = -1.f;
#pragma unroll
    for (int e = 0; e < E; ++e) {
      if (e == i0) continue;
      if (p[e] > p1) { p1 = p[e]; i1 = e; }
    }
    top_ids[token * 2 + 0] = i0;
    top_ids[token * 2 + 1] = i1;
    top_w[token * 2 + 0] = p0 * inv;   // softmax probs, top-k, no renorm
    top_w[token * 2 + 1] = p1 * inv;
  }
}

// ------------------------------------------------------------ build lists ---
__launch_bounds__(256)
__global__ void build_lists_kernel(const int* __restrict__ top_ids,
                                   const float* __restrict__ top_w,
                                   int* __restrict__ list_tok,
                                   float* __restrict__ slot_w,
                                   int* __restrict__ pos,
                                   int* __restrict__ counts,
                                   int* __restrict__ bases) {
  __shared__ int sCnt[E], sBase[E];
  const int e = threadIdx.x >> 5;     // one wave per expert (8 waves)
  const int lane = threadIdx.x & 31;

  int cnt = 0;
  for (int t0 = 0; t0 < T; t0 += 32) {
    const int t = t0 + lane;
    const bool f = (top_ids[t * 2] == e) || (top_ids[t * 2 + 1] == e);
    cnt += __popcll(__ballot(f));
  }
  if (lane == 0) sCnt[e] = cnt;
  __syncthreads();
  if (threadIdx.x == 0) {
    int b = 0;
    for (int i = 0; i < E; ++i) { sBase[i] = b; b += sCnt[i]; }
  }
  __syncthreads();
  const int base = sBase[e];

  int off = 0;
  for (int t0 = 0; t0 < T; t0 += 32) {
    const int t = t0 + lane;
    const int id0 = top_ids[t * 2], id1 = top_ids[t * 2 + 1];
    const bool f = (id0 == e) || (id1 == e);
    const unsigned long long m = __ballot(f);
    const int pre = __popcll(m & ((1ull << lane) - 1ull));
    if (f) {
      const int slot = base + off + pre;
      const int which = (id0 == e) ? 0 : 1;
      list_tok[slot] = t;
      slot_w[slot] = top_w[t * 2 + which];
      pos[t * 2 + which] = slot;
    }
    off += __popcll(m);
  }
  if (lane == 0) { counts[e] = cnt; bases[e] = base; }
}

// ------------------------------------------------------- fragment helpers ---
__device__ __forceinline__ void cvt_store16(__bf16* dst, float4 a, float4 b,
                                            float4 c, float4 d) {
  v16bf v;
  v[0]  = (__bf16)a.x; v[1]  = (__bf16)a.y; v[2]  = (__bf16)a.z; v[3]  = (__bf16)a.w;
  v[4]  = (__bf16)b.x; v[5]  = (__bf16)b.y; v[6]  = (__bf16)b.z; v[7]  = (__bf16)b.w;
  v[8]  = (__bf16)c.x; v[9]  = (__bf16)c.y; v[10] = (__bf16)c.z; v[11] = (__bf16)c.w;
  v[12] = (__bf16)d.x; v[13] = (__bf16)d.y; v[14] = (__bf16)d.z; v[15] = (__bf16)d.w;
  *(v16bf*)dst = v;
}

// ------------------------------------------------- GEMM1: x@ws.T + SwiGLU ---
// C_gate/C_up tile: 128 tokens x 128 cols, K = H = 2048, bf16 WMMA.
__launch_bounds__(256)
__global__ void moe_gemm1(const float* __restrict__ x,
                          const float* __restrict__ ws,
                          const int* __restrict__ list_tok,
                          const int* __restrict__ counts,
                          const int* __restrict__ bases,
                          __bf16* __restrict__ hbuf) {
  const int e = blockIdx.z;
  const int cnt = counts[e];
  const int tile_m = blockIdx.y;
  if (tile_m * 128 >= cnt) return;
  const int base = bases[e];
  const int tile_n = blockIdx.x;

  __shared__ alignas(64) __bf16 sA[2][4096];   // 128x32 (8 frags of 16x32)
  __shared__ alignas(64) __bf16 sBg[2][4096];  // 32x128 gate
  __shared__ alignas(64) __bf16 sBu[2][4096];  // 32x128 up

  const int tid = threadIdx.x;
  const int sub = tid >> 5;          // staging fragment / wave id, 0..7
  const int lane = tid & 31;
  const int l15 = lane & 15;
  const int lhi = lane >> 4;         // 0/1
  const int mw = sub >> 2;           // wave M group (0..1)
  const int nw = sub & 3;            // wave N group (0..3)

  // A source: gathered token row; ISA A layout: K = (e&7)+((e>>3)<<4)+(lhi<<3)
  const int mrow = tile_m * 128 + sub * 16 + l15;
  const int arow = min(mrow, cnt - 1);
  const float* xrow = x + (size_t)list_tok[base + arow] * H + (lhi << 3);
  // B source: ws rows (gate = n, up = I + n); B layout: K = (e&7)+((e>>3)<<4)+(lhi<<3)
  const int ncol = tile_n * 128 + sub * 16 + l15;
  const float* wgrow = ws + (size_t)e * TWOI * H + (size_t)ncol * H + (lhi << 3);
  const float* wurow = ws + (size_t)e * TWOI * H + (size_t)(I + ncol) * H + (lhi << 3);

  const v8f vzero = {0.f, 0.f, 0.f, 0.f, 0.f, 0.f, 0.f, 0.f};
  v8f accg[4][2], accu[4][2];
#pragma unroll
  for (int i = 0; i < 4; ++i)
#pragma unroll
    for (int j = 0; j < 2; ++j) { accg[i][j] = vzero; accu[i][j] = vzero; }

  float4 ra0, ra1, ra2, ra3, rg0, rg1, rg2, rg3, ru0, ru1, ru2, ru3;
  // prologue loads (K-step 0)
  ra0 = *(const float4*)(xrow + 0);  ra1 = *(const float4*)(xrow + 4);
  ra2 = *(const float4*)(xrow + 16); ra3 = *(const float4*)(xrow + 20);
  rg0 = *(const float4*)(wgrow + 0); rg1 = *(const float4*)(wgrow + 4);
  rg2 = *(const float4*)(wgrow + 16); rg3 = *(const float4*)(wgrow + 20);
  ru0 = *(const float4*)(wurow + 0); ru1 = *(const float4*)(wurow + 4);
  ru2 = *(const float4*)(wurow + 16); ru3 = *(const float4*)(wurow + 20);

  const int KT = H / 32;  // 64
#pragma unroll 1
  for (int kk = 0; kk < KT; ++kk) {
    const int buf = kk & 1;
    __bf16* da = &sA[buf][sub * 512 + lane * 16];
    cvt_store16(da, ra0, ra1, ra2, ra3);
    cvt_store16(&sBg[buf][sub * 512 + lane * 16], rg0, rg1, rg2, rg3);
    cvt_store16(&sBu[buf][sub * 512 + lane * 16], ru0, ru1, ru2, ru3);
    __syncthreads();

    if (kk + 1 < KT) {
      const int kb = (kk + 1) * 32;
      ra0 = *(const float4*)(xrow + kb + 0);  ra1 = *(const float4*)(xrow + kb + 4);
      ra2 = *(const float4*)(xrow + kb + 16); ra3 = *(const float4*)(xrow + kb + 20);
      rg0 = *(const float4*)(wgrow + kb + 0); rg1 = *(const float4*)(wgrow + kb + 4);
      rg2 = *(const float4*)(wgrow + kb + 16); rg3 = *(const float4*)(wgrow + kb + 20);
      ru0 = *(const float4*)(wurow + kb + 0); ru1 = *(const float4*)(wurow + kb + 4);
      ru2 = *(const float4*)(wurow + kb + 16); ru3 = *(const float4*)(wurow + kb + 20);
      if (kk + 2 < KT) {
        __builtin_prefetch((const void*)(wgrow + (kk + 2) * 32), 0, 1);
        __builtin_prefetch((const void*)(wurow + (kk + 2) * 32), 0, 1);
      }
    }

    v16bf af[4];
#pragma unroll
    for (int fm = 0; fm < 4; ++fm)
      af[fm] = *(const v16bf*)&sA[buf][(mw * 4 + fm) * 512 + lane * 16];
#pragma unroll
    for (int fn = 0; fn < 2; ++fn) {
      const v16bf bg = *(const v16bf*)&sBg[buf][(nw * 2 + fn) * 512 + lane * 16];
      const v16bf bu = *(const v16bf*)&sBu[buf][(nw * 2 + fn) * 512 + lane * 16];
#pragma unroll
      for (int fm = 0; fm < 4; ++fm) {
        accg[fm][fn] = __builtin_amdgcn_wmma_f32_16x16x32_bf16(
            false, af[fm], false, bg, (short)0, accg[fm][fn], false, false);
        accu[fm][fn] = __builtin_amdgcn_wmma_f32_16x16x32_bf16(
            false, af[fm], false, bu, (short)0, accu[fm][fn], false, false);
      }
    }
  }

  // epilogue: h = silu(gate) * up, bf16, guarded scatter to compact slots
#pragma unroll
  for (int fm = 0; fm < 4; ++fm) {
    const int r0 = tile_m * 128 + mw * 64 + fm * 16 + (lhi << 3);
#pragma unroll
    for (int fn = 0; fn < 2; ++fn) {
      const int N = tile_n * 128 + nw * 32 + fn * 16 + l15;
#pragma unroll
      for (int r = 0; r < 8; ++r) {
        const int row = r0 + r;
        if (row < cnt) {
          const float gv = accg[fm][fn][r];
          const float uv = accu[fm][fn][r];
          const float hv = (gv / (1.f + __expf(-gv))) * uv;
          hbuf[(size_t)(base + row) * I + N] = (__bf16)hv;
        }
      }
    }
  }
}

// --------------------------------------------- GEMM2: h@w2s.T, scale by w ---
__launch_bounds__(256)
__global__ void moe_gemm2(const __bf16* __restrict__ hbuf,
                          const float* __restrict__ w2s,
                          const float* __restrict__ slot_w,
                          const int* __restrict__ counts,
                          const int* __restrict__ bases,
                          float* __restrict__ Y) {
  const int e = blockIdx.z;
  const int cnt = counts[e];
  const int tile_m = blockIdx.y;
  if (tile_m * 128 >= cnt) return;
  const int base = bases[e];
  const int tile_n = blockIdx.x;

  __shared__ alignas(64) __bf16 sA[2][4096];
  __shared__ alignas(64) __bf16 sB[2][4096];

  const int tid = threadIdx.x;
  const int sub = tid >> 5;
  const int lane = tid & 31;
  const int l15 = lane & 15;
  const int lhi = lane >> 4;
  const int mw = sub >> 2;
  const int nw = sub & 3;

  const int mrow = tile_m * 128 + sub * 16 + l15;
  const int arow = min(mrow, cnt - 1);
  const __bf16* hrow = hbuf + (size_t)(base + arow) * I + (lhi << 3);
  const int ncol = tile_n * 128 + sub * 16 + l15;
  const float* w2row = w2s + (size_t)e * H * I + (size_t)ncol * I + (lhi << 3);

  const v8f vzero = {0.f, 0.f, 0.f, 0.f, 0.f, 0.f, 0.f, 0.f};
  v8f acc[4][2];
#pragma unroll
  for (int i = 0; i < 4; ++i)
#pragma unroll
    for (int j = 0; j < 2; ++j) acc[i][j] = vzero;

  float4 ra0, ra1, rb0, rb1, rb2, rb3;
  ra0 = *(const float4*)(hrow + 0);      // 8 bf16 = 16 B
  ra1 = *(const float4*)(hrow + 16);
  rb0 = *(const float4*)(w2row + 0); rb1 = *(const float4*)(w2row + 4);
  rb2 = *(const float4*)(w2row + 16); rb3 = *(const float4*)(w2row + 20);

  const int KT = I / 32;  // 128
#pragma unroll 1
  for (int kk = 0; kk < KT; ++kk) {
    const int buf = kk & 1;
    float4* da = (float4*)&sA[buf][sub * 512 + lane * 16];
    da[0] = ra0; da[1] = ra1;                          // already bf16
    cvt_store16(&sB[buf][sub * 512 + lane * 16], rb0, rb1, rb2, rb3);
    __syncthreads();

    if (kk + 1 < KT) {
      const int kb = (kk + 1) * 32;
      ra0 = *(const float4*)(hrow + kb);
      ra1 = *(const float4*)(hrow + kb + 16);
      rb0 = *(const float4*)(w2row + kb + 0); rb1 = *(const float4*)(w2row + kb + 4);
      rb2 = *(const float4*)(w2row + kb + 16); rb3 = *(const float4*)(w2row + kb + 20);
      if (kk + 2 < KT)
        __builtin_prefetch((const void*)(w2row + (kk + 2) * 32), 0, 1);
    }

    v16bf af[4];
#pragma unroll
    for (int fm = 0; fm < 4; ++fm)
      af[fm] = *(const v16bf*)&sA[buf][(mw * 4 + fm) * 512 + lane * 16];
#pragma unroll
    for (int fn = 0; fn < 2; ++fn) {
      const v16bf bv = *(const v16bf*)&sB[buf][(nw * 2 + fn) * 512 + lane * 16];
#pragma unroll
      for (int fm = 0; fm < 4; ++fm)
        acc[fm][fn] = __builtin_amdgcn_wmma_f32_16x16x32_bf16(
            false, af[fm], false, bv, (short)0, acc[fm][fn], false, false);
    }
  }

#pragma unroll
  for (int fm = 0; fm < 4; ++fm) {
    const int r0 = tile_m * 128 + mw * 64 + fm * 16 + (lhi << 3);
#pragma unroll
    for (int fn = 0; fn < 2; ++fn) {
      const int N = tile_n * 128 + nw * 32 + fn * 16 + l15;
#pragma unroll
      for (int r = 0; r < 8; ++r) {
        const int row = r0 + r;
        if (row < cnt) {
          const float sw = slot_w[base + row];
          Y[(size_t)(base + row) * H + N] = acc[fm][fn][r] * sw;
        }
      }
    }
  }
}

// --------------------------------------------------------------- combine ----
__launch_bounds__(256)
__global__ void combine_kernel(const float* __restrict__ Y,
                               const int* __restrict__ pos,
                               float* __restrict__ out) {
  const int t = blockIdx.y;
  const int c = (blockIdx.x * blockDim.x + threadIdx.x) * 4;
  const float4 a = *(const float4*)(Y + (size_t)pos[t * 2 + 0] * H + c);
  const float4 b = *(const float4*)(Y + (size_t)pos[t * 2 + 1] * H + c);
  const float4 o = {a.x + b.x, a.y + b.y, a.z + b.z, a.w + b.w};
  *(float4*)(out + (size_t)t * H + c) = o;
}

// ---------------------------------------------------------------- launch ----
extern "C" void kernel_launch(void* const* d_in, const int* in_sizes, int n_in,
                              void* d_out, int out_size, void* d_ws, size_t ws_size,
                              hipStream_t stream) {
  (void)in_sizes; (void)n_in; (void)out_size; (void)ws_size;
  const float* x   = (const float*)d_in[0];
  const float* wr  = (const float*)d_in[1];
  const float* ws  = (const float*)d_in[2];
  const float* w2s = (const float*)d_in[3];
  float* out = (float*)d_out;

  char* wsb = (char*)d_ws;
  int*    top_ids  = (int*)(wsb + OFF_TOPID);
  float*  top_w    = (float*)(wsb + OFF_TOPW);
  int*    list_tok = (int*)(wsb + OFF_LIST);
  float*  slot_w   = (float*)(wsb + OFF_SLOTW);
  int*    pos      = (int*)(wsb + OFF_POS);
  int*    counts   = (int*)(wsb + OFF_CNT);
  int*    bases    = (int*)(wsb + OFF_BASE);
  __bf16* hbuf     = (__bf16*)(wsb + OFF_H);
  float*  Y        = (float*)(wsb + OFF_Y);

  router_kernel<<<T / 8, 256, 0, stream>>>(x, wr, top_ids, top_w);
  build_lists_kernel<<<1, 256, 0, stream>>>(top_ids, top_w, list_tok, slot_w,
                                            pos, counts, bases);
  moe_gemm1<<<dim3(I / 128, T / 128, E), 256, 0, stream>>>(
      x, ws, list_tok, counts, bases, hbuf);
  moe_gemm2<<<dim3(H / 128, T / 128, E), 256, 0, stream>>>(
      hbuf, w2s, slot_w, counts, bases, Y);
  combine_kernel<<<dim3(H / 1024, T), 256, 0, stream>>>(Y, pos, out);
}
